// LSTMDecoder_12661563589156
// MI455X (gfx1250) — matrix operand order repaired and checked by reading, not compiled, not run on hardware
//
#include <hip/hip_runtime.h>
#include <hip/hip_bf16.h>
#include <math.h>

// ---------------------------------------------------------------------------
// Problem constants (match the reference)
// ---------------------------------------------------------------------------
#define HID   768
#define BATCH 1024
#define SEQ   256
#define H4    3072   // 4*HID
#define NC1   202
#define NC2   183
#define NC3   11

typedef __attribute__((ext_vector_type(16))) __bf16       v16bf;
typedef __attribute__((ext_vector_type(8)))  float        v8f;
typedef __attribute__((ext_vector_type(8)))  unsigned int v8u;

// fp32 -> bf16 with round-to-nearest-even (bit trick)
__device__ __forceinline__ unsigned f2bf(float f) {
    unsigned u = __float_as_uint(f);
    u += 0x7fffu + ((u >> 16) & 1u);
    return u >> 16;
}

__device__ __forceinline__ float wred_sum(float v) {
#pragma unroll
    for (int o = 16; o > 0; o >>= 1) v += __shfl_xor(v, o, 32);
    return v;
}
__device__ __forceinline__ float wred_max(float v) {
#pragma unroll
    for (int o = 16; o > 0; o >>= 1) v = fmaxf(v, __shfl_xor(v, o, 32));
    return v;
}

// ---------------------------------------------------------------------------
// GEMM staging helpers. A 128x32 fp32 tile is fetched with float4 loads into
// registers (overlapping WMMA of the previous chunk), then stored to LDS as
// bf16 pairs packed in dwords. LDS row stride = 17 dwords (34 bf16) keeps the
// 16-lane fragment gathers bank-conflict-free (gcd(17,64)=1).
// ---------------------------------------------------------------------------
__device__ __forceinline__ void load_regs(const float* __restrict__ P,
                                          int rowBase, int rowMax, int ld,
                                          int k0, int tid, float4 r[4]) {
#pragma unroll
    for (int i = 0; i < 4; i++) {
        int idx = tid + i * 256;      // 1024 float4 loads cover 128x32
        int row = idx >> 3;           // 8 float4 per 32-float row
        int c4  = idx & 7;
        int rr  = rowBase + row;
        if (rr < rowMax)
            r[i] = *(const float4*)(P + (size_t)rr * ld + k0 + c4 * 4);
        else
            r[i] = make_float4(0.f, 0.f, 0.f, 0.f);
    }
}

__device__ __forceinline__ void store_lds(const float4 r[4], unsigned* lds,
                                          int tid) {
#pragma unroll
    for (int i = 0; i < 4; i++) {
        int idx = tid + i * 256;
        int row = idx >> 3;
        int c4  = idx & 7;
        lds[row * 17 + c4 * 2]     = f2bf(r[i].x) | (f2bf(r[i].y) << 16);
        lds[row * 17 + c4 * 2 + 1] = f2bf(r[i].z) | (f2bf(r[i].w) << 16);
    }
}

// One BK=32 step: each wave does 4x2 WMMA tiles (64x32 of the 128x128 block).
// Fragment layouts per CDNA5 ISA 7.12.2:
//   A (16x32 bf16): lane m=lane&15, half=lane>>4; dword q<4 -> K=half*8+2q,
//                   q>=4 -> K=16+half*8+2(q-4)
//   B (32x16 bf16): lane n=lane&15; same K mapping as A
__device__ __forceinline__ void mma_step(const unsigned* AsU, const unsigned* WsU,
                                         int wm, int wn, int half, int mr,
                                         v8f acc[4][2]) {
    v16bf bfr[2];
#pragma unroll
    for (int j = 0; j < 2; j++) {
        const unsigned* bp = &WsU[(wn * 32 + j * 16 + mr) * 17];
        v8u bu;
#pragma unroll
        for (int q = 0; q < 8; q++) bu[q] = bp[(q >> 2) * 8 + half * 4 + (q & 3)];
        bfr[j] = __builtin_bit_cast(v16bf, bu);
    }
#pragma unroll
    for (int i = 0; i < 4; i++) {
        const unsigned* ap = &AsU[(wm * 64 + i * 16 + mr) * 17];
        v8u au;
#pragma unroll
        for (int q = 0; q < 4; q++) au[q] = ap[half * 4 + q];
#pragma unroll
        for (int q = 0; q < 4; q++) au[4 + q] = ap[8 + half * 4 + q];
        v16bf afr = __builtin_bit_cast(v16bf, au);
#pragma unroll
        for (int j = 0; j < 2; j++)
            acc[i][j] = __builtin_amdgcn_wmma_f32_16x16x32_bf16(
                false, afr, false, bfr[j], (short)0, acc[i][j], false, false);
    }
}

// ---------------------------------------------------------------------------
// C[M,N] = act( A[M,K] @ W[N,K]^T  (+ A2[M,K] @ W2[N,K]^T)
//               + bias1[N] + bias2[N] + addmat[M,N] )
// act = relu if relu!=0. 256 threads = 8 waves, block tile 128x128, BK=32.
// Double-buffered LDS ping-pong: global float4 loads for chunk c+1 are issued
// before the WMMAs of chunk c, hiding VMEM latency behind the matrix pipe.
// One __syncthreads per chunk (barrier at chunk c orders all LDS reads of
// chunk c-1 before the buffer is overwritten at chunk c+1).
// ---------------------------------------------------------------------------
__global__ __launch_bounds__(256)
void gemm_wmma_bf16(const float* __restrict__ A,  const float* __restrict__ W,
                    const float* __restrict__ A2, const float* __restrict__ W2,
                    const float* __restrict__ bias1, const float* __restrict__ bias2,
                    const float* __restrict__ addmat,
                    float* __restrict__ C, int M, int N, int K, int relu) {
    __shared__ unsigned AsU[2][128 * 17];
    __shared__ unsigned WsU[2][128 * 17];

    const int tid    = threadIdx.x;
    const int blockN = blockIdx.x * 128;
    const int blockM = blockIdx.y * 128;
    const int wave   = tid >> 5;
    const int lane   = tid & 31;
    const int wm     = wave >> 2;   // 0..1  (M direction)
    const int wn     = wave & 3;    // 0..3  (N direction)
    const int half   = lane >> 4;
    const int mr     = lane & 15;
    const bool dual  = (A2 != nullptr);

    v8f acc[4][2];
    const v8f vzero = {0.f, 0.f, 0.f, 0.f, 0.f, 0.f, 0.f, 0.f};
#pragma unroll
    for (int i = 0; i < 4; i++)
#pragma unroll
        for (int j = 0; j < 2; j++) acc[i][j] = vzero;

    // Virtual chunk list: dual interleaves (A,W) and (A2,W2) per k0.
    const int nk      = K >> 5;
    const int nchunks = dual ? (nk << 1) : nk;

    float4 ra[4], rw[4];
    load_regs(A, blockM, M, K, 0, tid, ra);
    load_regs(W, blockN, N, K, 0, tid, rw);

    int p = 0;
    for (int c = 0; c < nchunks; ++c) {
        store_lds(ra, AsU[p], tid);
        store_lds(rw, WsU[p], tid);
        __syncthreads();
        if (c + 1 < nchunks) {
            const int cn = c + 1;
            const float* Ac = dual ? ((cn & 1) ? A2 : A) : A;
            const float* Wc = dual ? ((cn & 1) ? W2 : W) : W;
            const int k0    = dual ? ((cn >> 1) << 5) : (cn << 5);
            load_regs(Ac, blockM, M, K, k0, tid, ra);
            load_regs(Wc, blockN, N, K, k0, tid, rw);
        }
        mma_step(AsU[p], WsU[p], wm, wn, half, mr, acc);
        p ^= 1;
    }

    // Epilogue. C layout: VGPR r -> M = r + 8*half, N = lane&15.
#pragma unroll
    for (int j = 0; j < 2; j++) {
        int n = blockN + wn * 32 + j * 16 + mr;
        if (n >= N) continue;
        float bv = 0.f;
        if (bias1) bv += bias1[n];
        if (bias2) bv += bias2[n];
#pragma unroll
        for (int i = 0; i < 4; i++) {
#pragma unroll
            for (int r = 0; r < 8; r++) {
                int m = blockM + wm * 64 + i * 16 + half * 8 + r;
                if (m < M) {
                    float v = acc[i][j][r] + bv;
                    if (addmat) v += addmat[(size_t)m * N + n];
                    if (relu) v = fmaxf(v, 0.f);
                    C[(size_t)m * N + n] = v;
                }
            }
        }
    }
}

// ---------------------------------------------------------------------------
// LSTM gate pointwise: gates G[B, 4H] in torch order (i, f, g, o)
// c_new = sig(f)*c_prev + sig(i)*tanh(g);  h_new = sig(o)*tanh(c_new)
// ---------------------------------------------------------------------------
__global__ __launch_bounds__(256)
void lstm_pointwise(const float* __restrict__ G, const float* __restrict__ cprev,
                    float* __restrict__ h, float* __restrict__ c) {
    int idx = blockIdx.x * 256 + threadIdx.x;
    if (idx >= BATCH * HID) return;
    int b = idx / HID, j = idx % HID;
    const float* g = G + (size_t)b * H4;
    float ig = 1.f / (1.f + __expf(-g[j]));
    float fg = 1.f / (1.f + __expf(-g[HID + j]));
    float gg = tanhf(g[2 * HID + j]);
    float og = 1.f / (1.f + __expf(-g[3 * HID + j]));
    float cp = cprev ? cprev[idx] : 0.f;
    float cn = fg * cp + ig * gg;
    c[idx] = cn;
    h[idx] = og * tanhf(cn);
}

// ---------------------------------------------------------------------------
// Attention pass 1: sc[t][b][s] = attn[b,s,:] . h_t[b,:]  for t in {1,2,3}
// One block per batch element; one wave per score row; float4 per-lane loads
// (512 B contiguous per wave step); single read of attn for all 3 tasks.
// ---------------------------------------------------------------------------
__global__ __launch_bounds__(256)
void attn_scores(const float* __restrict__ attn, const float* __restrict__ h1,
                 const float* __restrict__ h2, const float* __restrict__ h3,
                 float* __restrict__ sc) {
    __shared__ __align__(16) float hh[3][HID];
    const int b = blockIdx.x, tid = threadIdx.x;
    for (int i = tid; i < HID; i += 256) {
        hh[0][i] = h1[(size_t)b * HID + i];
        hh[1][i] = h2[(size_t)b * HID + i];
        hh[2][i] = h3[(size_t)b * HID + i];
    }
    __syncthreads();
    const int wave = tid >> 5, lane = tid & 31;
    const float4* hh0 = (const float4*)hh[0];
    const float4* hh1 = (const float4*)hh[1];
    const float4* hh2 = (const float4*)hh[2];
    for (int s = wave; s < SEQ; s += 8) {
        const float4* row = (const float4*)(attn + ((size_t)b * SEQ + s) * HID);
        float d0 = 0.f, d1 = 0.f, d2 = 0.f;
#pragma unroll
        for (int j = 0; j < HID / 128; j++) {   // 6 float4 steps per lane
            int k4 = lane + j * 32;
            float4 v = row[k4];
            float4 a0 = hh0[k4], a1 = hh1[k4], a2 = hh2[k4];
            d0 += v.x * a0.x + v.y * a0.y + v.z * a0.z + v.w * a0.w;
            d1 += v.x * a1.x + v.y * a1.y + v.z * a1.z + v.w * a1.w;
            d2 += v.x * a2.x + v.y * a2.y + v.z * a2.z + v.w * a2.w;
        }
        d0 = wred_sum(d0); d1 = wred_sum(d1); d2 = wred_sum(d2);
        if (lane == 0) {
            sc[(size_t)0 * BATCH * SEQ + (size_t)b * SEQ + s] = d0;
            sc[(size_t)1 * BATCH * SEQ + (size_t)b * SEQ + s] = d1;
            sc[(size_t)2 * BATCH * SEQ + (size_t)b * SEQ + s] = d2;
        }
    }
}

// ---------------------------------------------------------------------------
// Attention pass 2: softmax the 3 score rows of this batch element in LDS,
// then stream attn[b] once, accumulating a_t[b,:] = sum_s w_t[s]*attn[b,s,:]
// for all three tasks simultaneously (thread owns 3 H-columns).
// ---------------------------------------------------------------------------
__global__ __launch_bounds__(256)
void attn_apply(const float* __restrict__ attn, const float* __restrict__ sc,
                float* __restrict__ a1, float* __restrict__ a2,
                float* __restrict__ a3) {
    __shared__ float w[3][SEQ];
    const int b = blockIdx.x, tid = threadIdx.x;
#pragma unroll
    for (int t = 0; t < 3; t++)
        w[t][tid] = sc[(size_t)t * BATCH * SEQ + (size_t)b * SEQ + tid];
    __syncthreads();
    const int wave = tid >> 5, lane = tid & 31;
    if (wave < 3) {  // one wave per task softmax
        float mx = -3.402823466e38f;
        for (int i = lane; i < SEQ; i += 32) mx = fmaxf(mx, w[wave][i]);
        mx = wred_max(mx);
        float sm = 0.f;
        for (int i = lane; i < SEQ; i += 32) {
            float e = __expf(w[wave][i] - mx);
            w[wave][i] = e;
            sm += e;
        }
        sm = wred_sum(sm);
        float inv = 1.f / sm;
        for (int i = lane; i < SEQ; i += 32) w[wave][i] *= inv;
    }
    __syncthreads();

    float acc00 = 0.f, acc01 = 0.f, acc02 = 0.f;
    float acc10 = 0.f, acc11 = 0.f, acc12 = 0.f;
    float acc20 = 0.f, acc21 = 0.f, acc22 = 0.f;
    for (int s = 0; s < SEQ; s++) {
        const float* row = attn + ((size_t)b * SEQ + s) * HID;
        float v0 = row[tid], v1 = row[tid + 256], v2 = row[tid + 512];
        float w0 = w[0][s], w1 = w[1][s], w2 = w[2][s];
        acc00 += w0 * v0; acc01 += w0 * v1; acc02 += w0 * v2;
        acc10 += w1 * v0; acc11 += w1 * v1; acc12 += w1 * v2;
        acc20 += w2 * v0; acc21 += w2 * v1; acc22 += w2 * v2;
    }
    a1[(size_t)b * HID + tid]       = acc00;
    a1[(size_t)b * HID + tid + 256] = acc01;
    a1[(size_t)b * HID + tid + 512] = acc02;
    a2[(size_t)b * HID + tid]       = acc10;
    a2[(size_t)b * HID + tid + 256] = acc11;
    a2[(size_t)b * HID + tid + 512] = acc12;
    a3[(size_t)b * HID + tid]       = acc20;
    a3[(size_t)b * HID + tid + 256] = acc21;
    a3[(size_t)b * HID + tid + 512] = acc22;
}

// ---------------------------------------------------------------------------
// Host-side orchestration
// ---------------------------------------------------------------------------
extern "C" void kernel_launch(void* const* d_in, const int* in_sizes, int n_in,
                              void* d_out, int out_size, void* d_ws, size_t ws_size,
                              hipStream_t stream) {
    const float* x    = (const float*)d_in[0];
    const float* attn = (const float*)d_in[1];
    const float* Wih  = (const float*)d_in[2];
    const float* Whh  = (const float*)d_in[3];
    const float* bih  = (const float*)d_in[4];
    const float* bhh  = (const float*)d_in[5];
    const float* hfcW = (const float*)d_in[6];
    const float* hfcB = (const float*)d_in[7];
    const float* cfcW = (const float*)d_in[8];
    const float* cfcB = (const float*)d_in[9];
    const float* midW = (const float*)d_in[10];
    const float* midB = (const float*)d_in[11];
    const float* o1W  = (const float*)d_in[12];
    const float* o1B  = (const float*)d_in[13];
    const float* o2W  = (const float*)d_in[14];
    const float* o2B  = (const float*)d_in[15];
    const float* o3W  = (const float*)d_in[16];
    const float* o3B  = (const float*)d_in[17];
    float* out = (float*)d_out;

    // Workspace layout (floats): ~59.8 MB total
    const size_t BH = (size_t)BATCH * HID;
    float* ws  = (float*)d_ws;
    float* G   = ws;                              // BATCH*H4
    float* h1  = G + (size_t)BATCH * H4;
    float* c1  = h1 + BH;
    float* h2  = c1 + BH;
    float* c2  = h2 + BH;
    float* h3  = c2 + BH;
    float* c3  = h3 + BH;
    float* hp3 = c3 + BH;
    float* cp3 = hp3 + BH;
    float* sc  = cp3 + BH;                        // 3*BATCH*SEQ
    float* a1  = sc + (size_t)3 * BATCH * SEQ;
    float* a2  = a1 + BH;
    float* a3  = a2 + BH;
    float* m1  = a3 + BH;
    float* m2  = m1 + BH;
    float* m3  = m2 + BH;

    dim3 blk(256);
    auto gemm = [&](const float* A, const float* W, const float* A2,
                    const float* W2, const float* b1, const float* b2,
                    const float* addm, float* Cc, int M, int N, int K, int relu) {
        dim3 grid((N + 127) / 128, (M + 127) / 128);
        gemm_wmma_bf16<<<grid, blk, 0, stream>>>(A, W, A2, W2, b1, b2, addm,
                                                 Cc, M, N, K, relu);
    };
    const int pwGrid = (BATCH * HID + 255) / 256;

    // Task 1 (crit): h_prev = c_prev = 0  -> gates = x@Wih0^T + bih0 + bhh0
    gemm(x, Wih, nullptr, nullptr, bih, bhh, nullptr, G, BATCH, H4, HID, 0);
    lstm_pointwise<<<pwGrid, blk, 0, stream>>>(G, nullptr, h1, c1);

    // Task 2 (law): state = (h1, c1); fused dual GEMM x@Wih1^T + h1@Whh1^T
    gemm(x, Wih + (size_t)H4 * HID, h1, Whh + (size_t)H4 * HID,
         bih + H4, bhh + H4, nullptr, G, BATCH, H4, HID, 0);
    lstm_pointwise<<<pwGrid, blk, 0, stream>>>(G, c1, h2, c2);

    // Edge 2->3: hp3 = h1 + h2@hfcW^T + hfcB ; cp3 = c1 + c2@cfcW^T + cfcB
    gemm(h2, hfcW, nullptr, nullptr, hfcB, nullptr, h1, hp3, BATCH, HID, HID, 0);
    gemm(c2, cfcW, nullptr, nullptr, cfcB, nullptr, c1, cp3, BATCH, HID, HID, 0);

    // Task 3 (time): state = (hp3, cp3)
    gemm(x, Wih + 2 * (size_t)H4 * HID, hp3, Whh + 2 * (size_t)H4 * HID,
         bih + 2 * H4, bhh + 2 * H4, nullptr, G, BATCH, H4, HID, 0);
    lstm_pointwise<<<pwGrid, blk, 0, stream>>>(G, cp3, h3, c3);

    // Attention: 2 streaming passes over attn (fused across all 3 tasks)
    attn_scores<<<BATCH, blk, 0, stream>>>(attn, h1, h2, h3, sc);
    attn_apply<<<BATCH, blk, 0, stream>>>(attn, sc, a1, a2, a3);

    // Heads: relu(midfc) then outfc, written straight into concatenated d_out
    gemm(a1, midW, nullptr, nullptr, midB, nullptr, nullptr, m1, BATCH, HID, HID, 1);
    gemm(a2, midW + (size_t)HID * HID, nullptr, nullptr, midB + HID, nullptr,
         nullptr, m2, BATCH, HID, HID, 1);
    gemm(a3, midW + 2 * (size_t)HID * HID, nullptr, nullptr, midB + 2 * HID,
         nullptr, nullptr, m3, BATCH, HID, HID, 1);

    gemm(m1, o1W, nullptr, nullptr, o1B, nullptr, nullptr, out, BATCH, NC1, HID, 0);
    gemm(m2, o2W, nullptr, nullptr, o2B, nullptr, nullptr,
         out + (size_t)BATCH * NC1, BATCH, NC2, HID, 0);
    gemm(m3, o3W, nullptr, nullptr, o3B, nullptr, nullptr,
         out + (size_t)BATCH * (NC1 + NC2), BATCH, NC3, HID, 0);
}
